// StateLocationGraphGRUCell_59528246723079
// MI455X (gfx1250) — matrix operand order repaired and checked by reading, not compiled, not run on hardware
//
#include <hip/hip_runtime.h>
#include <math.h>

// ---------------------------------------------------------------------------
// StateLocationGraphGRUCell forward for MI455X (gfx1250, wave32, WMMA).
// BF16 WMMA (V_WMMA_F32_16X16X32_BF16) for all GEMMs + fused flash attention.
// Round 3: pure 32-bit bf16 pack math (no 16-bit sub-register moves).
// ---------------------------------------------------------------------------

typedef __attribute__((ext_vector_type(16))) __bf16 v16bf;
typedef __attribute__((ext_vector_type(8)))  float  v8f;

#define HDIM   256
#define DCAT   384
#define FFDIM  1024
#define SEQ    512
#define BATCH  64
#define NTOK   (BATCH * SEQ)   // 32768 tokens per half
#define NHEADS 4
#define DH     64

// ---- bf16 helpers (all 32-bit integer math, RNE rounding) -----------------
__device__ __forceinline__ unsigned int bfbits(float f) {
  union { float f; unsigned int u; } c; c.f = f;
  return c.u + 0x7FFFu + ((c.u >> 16) & 1u);   // rounded bf16 in bits [31:16]
}
__device__ __forceinline__ unsigned short f2bf(float f) {
  return (unsigned short)(bfbits(f) >> 16);
}
__device__ __forceinline__ unsigned int pack2bf(float lo, float hi) {
  return (bfbits(lo) >> 16) | (bfbits(hi) & 0xFFFF0000u);
}

union FragB16 { v16bf v; unsigned int u[8]; };

// A-matrix 16x32 bf16 fragment (ISA 7.12.2): lane m = lane&15, half = lane>>4.
// VGPR j<4 : K = 2j + 8*half ; j>=4 : K = 16 + 2(j-4) + 8*half (pairs).
__device__ __forceinline__ v16bf lds_load_a16(const unsigned short* p, int stride,
                                              int m, int hf) {
  FragB16 f;
  const unsigned short* row = p + m * stride;
#pragma unroll
  for (int j = 0; j < 8; ++j) {
    int kb = (j < 4) ? (2 * j + 8 * hf) : (16 + 2 * (j - 4) + 8 * hf);
    f.u[j] = *reinterpret_cast<const unsigned int*>(row + kb);
  }
  return f.v;
}

// B-matrix 32x16 bf16 fragment: lane n = lane&15; VGPR j pair K same as A (j<4: 2j+8*half; j>=4: 16+2(j-4)+8*half).
// LDS staged as [n][k] so each pair is one aligned 32-bit read.
__device__ __forceinline__ v16bf lds_load_b16(const unsigned short* p, int stride,
                                              int n, int hf) {
  FragB16 f;
  const unsigned short* row = p + n * stride;
#pragma unroll
  for (int j = 0; j < 8; ++j) {
    int kb = (j < 4) ? (2 * j + 8 * hf) : (16 + 2 * (j - 4) + 8 * hf);
    f.u[j] = *reinterpret_cast<const unsigned int*>(row + kb);
  }
  return f.v;
}

__device__ __forceinline__ v8f wmma_bf16(v16bf a, v16bf b, v8f c) {
  return __builtin_amdgcn_wmma_f32_16x16x32_bf16(false, a, false, b,
                                                 (short)0, c, false, false);
}

// Segmented token -> physical row:  phys = (tok/512)*segStride + tok%512 + rowOff
__device__ __forceinline__ int map_row(int r, int segStride, int rowOff) {
  return ((r >> 9) * segStride) + (r & 511) + rowOff;
}

// ---------------------------------------------------------------------------
// WMMA GEMM: C[M,N] = act(A[M,K] @ W[K,N] + bias)   (fp32 in/out)
// Block = 256 threads (8 waves), 128x128 tile, wave grid 4x2 (each 32x64),
// K step 32 -> 64 WMMA per block per step. act: 0=none 1=sigmoid 2=tanh
// ---------------------------------------------------------------------------
__global__ __launch_bounds__(256) void wmma_gemm_kernel(
    const float* __restrict__ A, const float* __restrict__ W,
    const float* __restrict__ bias, float* __restrict__ C,
    int M, int N, int K, int lda, int segStride, int rowOff, int act) {
  __shared__ unsigned short As[128 * 34];   // [m][k], pad 2
  __shared__ unsigned short Bs[128 * 34];   // [n][k], pad 2

  const int tid  = threadIdx.x;
  const int lane = tid & 31, wid = tid >> 5;
  const int hf   = lane >> 4, l15 = lane & 15;
  const int wm   = wid >> 1,  wn  = wid & 1;     // 4x2 wave grid
  const int rowBase = blockIdx.y * 128;
  const int colBase = blockIdx.x * 128;

  v8f acc[2][4] = {};
  for (int kk = 0; kk < K; kk += 32) {
    // stage A tile 128x32: 1024 float4 chunks, packed b32 LDS stores
#pragma unroll
    for (int i = 0; i < 4; ++i) {
      int idx = i * 256 + tid;
      int r = idx >> 3, c4 = (idx & 7) * 4;
      int prow = map_row(rowBase + r, segStride, rowOff);
      float4 v = *reinterpret_cast<const float4*>(&A[(size_t)prow * lda + kk + c4]);
      unsigned int* dst = reinterpret_cast<unsigned int*>(&As[r * 34 + c4]);
      dst[0] = pack2bf(v.x, v.y);
      dst[1] = pack2bf(v.z, v.w);
    }
    // stage W tile 32x128 transposed into Bs[n][k] (float4 along n)
#pragma unroll
    for (int i = 0; i < 4; ++i) {
      int idx = i * 256 + tid;
      int k = idx >> 5, n4 = (idx & 31) * 4;
      float4 v = *reinterpret_cast<const float4*>(&W[(size_t)(kk + k) * N + colBase + n4]);
      Bs[(n4 + 0) * 34 + k] = f2bf(v.x);
      Bs[(n4 + 1) * 34 + k] = f2bf(v.y);
      Bs[(n4 + 2) * 34 + k] = f2bf(v.z);
      Bs[(n4 + 3) * 34 + k] = f2bf(v.w);
    }
    if (kk + 32 < K)   // prefetch next W tile -> global_prefetch_b8
      __builtin_prefetch(&W[(size_t)(kk + 32 + (tid >> 5)) * N + colBase + (tid & 31) * 4], 0, 1);
    __syncthreads();

    v16bf a[2], b[4];
#pragma unroll
    for (int mi = 0; mi < 2; ++mi)
      a[mi] = lds_load_a16(As, 34, wm * 32 + mi * 16 + l15, hf);
#pragma unroll
    for (int ni = 0; ni < 4; ++ni)
      b[ni] = lds_load_b16(Bs, 34, wn * 64 + ni * 16 + l15, hf);
#pragma unroll
    for (int mi = 0; mi < 2; ++mi)
#pragma unroll
      for (int ni = 0; ni < 4; ++ni)
        acc[mi][ni] = wmma_bf16(a[mi], b[ni], acc[mi][ni]);
    __syncthreads();
  }

  // C fragment: VGPR r, lanes 0-15 -> M=r, lanes 16-31 -> M=8+r; N = lane&15
#pragma unroll
  for (int mi = 0; mi < 2; ++mi)
#pragma unroll
    for (int ni = 0; ni < 4; ++ni)
#pragma unroll
      for (int r = 0; r < 8; ++r) {
        int row = rowBase + wm * 32 + mi * 16 + r + 8 * hf;
        int col = colBase + wn * 64 + ni * 16 + l15;
        float v = acc[mi][ni][r] + bias[col];
        if (act == 1)      v = 1.0f / (1.0f + __expf(-v));
        else if (act == 2) v = tanhf(v);
        C[(size_t)row * N + col] = v;
      }
}

// ---------------------------------------------------------------------------
// Fused multihead attention, flash-style online softmax.
// Block = 256 threads (8 waves), one (batch, head, 128-query block).
// Each wave owns 16 queries; 32 keys per step; dh = 64.
// ---------------------------------------------------------------------------
__global__ __launch_bounds__(256) void attn_kernel(
    const float* __restrict__ Q, const float* __restrict__ Kt,
    const float* __restrict__ V, float* __restrict__ O) {
  __shared__ unsigned short Qs[128 * 66];       // [q][d]  (scaled bf16)
  __shared__ unsigned short Ks[32 * 66];        // [key][d]
  __shared__ unsigned short Vs[64 * 34];        // [d][key] (transposed)
  __shared__ unsigned short Ps[8][16 * 34];     // per-wave P tile [q][key]

  const int tid  = threadIdx.x;
  const int lane = tid & 31, wid = tid >> 5;
  const int hf   = lane >> 4, l15 = lane & 15;
  const int qBase = blockIdx.x * 128;
  const int head  = blockIdx.y;
  const int b     = blockIdx.z;
  const size_t tokBase = (size_t)b * SEQ;
  const int colOff = head * DH;

  // stage Q block (scale 1/sqrt(64) folded in): 2048 float4, packed stores
#pragma unroll
  for (int i = 0; i < 8; ++i) {
    int idx = i * 256 + tid;
    int qi = idx >> 4, d4 = (idx & 15) * 4;
    float4 v = *reinterpret_cast<const float4*>(
        &Q[(tokBase + qBase + qi) * HDIM + colOff + d4]);
    unsigned int* dst = reinterpret_cast<unsigned int*>(&Qs[qi * 66 + d4]);
    dst[0] = pack2bf(v.x * 0.125f, v.y * 0.125f);
    dst[1] = pack2bf(v.z * 0.125f, v.w * 0.125f);
  }

  v8f acc[4] = {};
  float mrow[8], lrow[8];
#pragma unroll
  for (int r = 0; r < 8; ++r) { mrow[r] = -1e30f; lrow[r] = 0.0f; }

  for (int kb = 0; kb < SEQ; kb += 32) {
    __syncthreads();
    // stage K (packed [key][d]) and V (transposed [d][key]) tiles, float4 loads
#pragma unroll
    for (int i = 0; i < 2; ++i) {
      int idx = i * 256 + tid;
      int kk = idx >> 4, d4 = (idx & 15) * 4;
      float4 kv = *reinterpret_cast<const float4*>(
          &Kt[(tokBase + kb + kk) * HDIM + colOff + d4]);
      float4 vv = *reinterpret_cast<const float4*>(
          &V[(tokBase + kb + kk) * HDIM + colOff + d4]);
      unsigned int* dk = reinterpret_cast<unsigned int*>(&Ks[kk * 66 + d4]);
      dk[0] = pack2bf(kv.x, kv.y);
      dk[1] = pack2bf(kv.z, kv.w);
      Vs[(d4 + 0) * 34 + kk] = f2bf(vv.x);
      Vs[(d4 + 1) * 34 + kk] = f2bf(vv.y);
      Vs[(d4 + 2) * 34 + kk] = f2bf(vv.z);
      Vs[(d4 + 3) * 34 + kk] = f2bf(vv.w);
    }
    __syncthreads();

    // scores 16x32: two 16x16 C frags, accumulate over dh in 2 K-steps
    v8f s0 = {}, s1 = {};
#pragma unroll
    for (int kt = 0; kt < 2; ++kt) {
      v16bf aq  = lds_load_a16(Qs + kt * 32, 66, wid * 16 + l15, hf);
      v16bf bk0 = lds_load_b16(Ks + kt * 32, 66, l15,            hf);
      v16bf bk1 = lds_load_b16(Ks + kt * 32, 66, 16 + l15,       hf);
      s0 = wmma_bf16(aq, bk0, s0);
      s1 = wmma_bf16(aq, bk1, s1);
    }

    // online softmax; C-frag row = r + 8*half, 16 lanes per half share a row
    unsigned short* pw = Ps[wid];
#pragma unroll
    for (int r = 0; r < 8; ++r) {
      float lm = fmaxf(s0[r], s1[r]);
#pragma unroll
      for (int ms = 1; ms < 16; ms <<= 1) lm = fmaxf(lm, __shfl_xor(lm, ms, 32));
      float mnew  = fmaxf(mrow[r], lm);
      float alpha = __expf(mrow[r] - mnew);
      float p0 = __expf(s0[r] - mnew);
      float p1 = __expf(s1[r] - mnew);
      float ps = p0 + p1;
#pragma unroll
      for (int ms = 1; ms < 16; ms <<= 1) ps += __shfl_xor(ps, ms, 32);
      lrow[r] = lrow[r] * alpha + ps;
      mrow[r] = mnew;
#pragma unroll
      for (int ni = 0; ni < 4; ++ni) acc[ni][r] *= alpha;
      int prow = r + 8 * hf;
      pw[prow * 34 + l15]      = f2bf(p0);   // C-layout -> LDS [q][key]
      pw[prow * 34 + 16 + l15] = f2bf(p1);
    }

    // acc += P(16x32) @ V(32x64): re-read P in A-layout (same-wave LDS, in order)
    v16bf ap = lds_load_a16(pw, 34, l15, hf);
#pragma unroll
    for (int ni = 0; ni < 4; ++ni) {
      v16bf bv = lds_load_b16(Vs, 34, ni * 16 + l15, hf);
      acc[ni] = wmma_bf16(ap, bv, acc[ni]);
    }
  }

  // epilogue: divide by row sums, write fp32
#pragma unroll
  for (int ni = 0; ni < 4; ++ni)
#pragma unroll
    for (int r = 0; r < 8; ++r) {
      int q = qBase + wid * 16 + r + 8 * hf;
      int d = ni * 16 + l15;
      O[(tokBase + q) * HDIM + colOff + d] = acc[ni][r] / lrow[r];
    }
}

// ---------------------------------------------------------------------------
// out = LayerNorm(x + res) * g + b     (one 256-wide row per block)
// ---------------------------------------------------------------------------
__global__ __launch_bounds__(256) void add_ln_kernel(
    const float* __restrict__ X, const float* __restrict__ R,
    const float* __restrict__ g, const float* __restrict__ bta,
    float* __restrict__ O) {
  __shared__ float red[256];
  const int row = blockIdx.x, tid = threadIdx.x;
  float v = X[(size_t)row * HDIM + tid] + R[(size_t)row * HDIM + tid];
  red[tid] = v; __syncthreads();
#pragma unroll
  for (int s = 128; s > 0; s >>= 1) { if (tid < s) red[tid] += red[tid + s]; __syncthreads(); }
  float mean = red[0] * (1.0f / HDIM);
  __syncthreads();
  float d = v - mean;
  red[tid] = d * d; __syncthreads();
#pragma unroll
  for (int s = 128; s > 0; s >>= 1) { if (tid < s) red[tid] += red[tid + s]; __syncthreads(); }
  float var = red[0] * (1.0f / HDIM);
  O[(size_t)row * HDIM + tid] = d * rsqrtf(var + 1e-5f) * g[tid] + bta[tid];
}

// ---------------------------------------------------------------------------
// Concat: O[row] = [ input_row(128) | (rgate? r*hx : hx)(256) ]
// ---------------------------------------------------------------------------
__global__ __launch_bounds__(256) void cat_kernel(
    const float* __restrict__ in_, const float* __restrict__ hx,
    const float* __restrict__ rgate, float* __restrict__ O, int srowOff) {
  const size_t idx = (size_t)blockIdx.x * 256 + threadIdx.x;  // NTOK*384
  const int row = (int)(idx / DCAT), c = (int)(idx % DCAT);
  const int prow = map_row(row, 1024, srowOff);
  float v;
  if (c < 128) {
    v = in_[(size_t)prow * 128 + c];
  } else {
    float h = hx[(size_t)prow * HDIM + (c - 128)];
    v = rgate ? rgate[(size_t)row * HDIM + (c - 128)] * h : h;
  }
  O[idx] = v;
}

// ---------------------------------------------------------------------------
// GRU blend: out = (1-z)*hx + z*h    -> writes directly into d_out slice
// ---------------------------------------------------------------------------
__global__ __launch_bounds__(256) void gru_kernel(
    const float* __restrict__ Z, const float* __restrict__ Hn,
    const float* __restrict__ hx, float* __restrict__ out, int srowOff) {
  const size_t idx = (size_t)blockIdx.x * 256 + threadIdx.x;  // NTOK*256
  const int row = (int)(idx >> 8), c = (int)(idx & 255);
  const int prow = map_row(row, 1024, srowOff);
  float z = Z[idx], h = Hn[idx];
  float hv = hx[(size_t)prow * HDIM + c];
  out[(size_t)prow * HDIM + c] = (1.0f - z) * hv + z * h;
}

// ---------------------------------------------------------------------------
// Host orchestration
// ---------------------------------------------------------------------------
extern "C" void kernel_launch(void* const* d_in, const int* in_sizes, int n_in,
                              void* d_out, int out_size, void* d_ws, size_t ws_size,
                              hipStream_t stream) {
  (void)in_sizes; (void)n_in; (void)out_size; (void)ws_size;

  const float* input_ = (const float*)d_in[0];   // (64,1024,128)
  const float* hx     = (const float*)d_in[1];   // (64,1024,256)
  const float* P[40];
  for (int i = 0; i < 40; ++i) P[i] = (const float*)d_in[2 + i];
  // params in setup_inputs() insertion order:
  enum {
    GUE_W = 0, GUE_B, GRE_W, GRE_B, GUL_W, GUL_B, GRL_W, GRL_B,
    PHE_W, PHE_B, PHL_W, PHL_B,
    AE_QW, AE_QB, AE_KW, AE_KB, AE_VW, AE_VB, AE_F1W, AE_F1B, AE_F2W, AE_F2B,
    AE_N1G, AE_N1B, AE_N2G, AE_N2B,
    AL_QW, AL_QB, AL_KW, AL_KB, AL_VW, AL_VB, AL_F1W, AL_F1B, AL_F2W, AL_F2B,
    AL_N1G, AL_N1B, AL_N2G, AL_N2B
  };

  float* out = (float*)d_out;                    // (64,1024,256)

  // workspace layout (fp32)
  float* WS  = (float*)d_ws;
  float* CAT = WS;                               // NTOK*384  (cat / gated x2)
  float* Zb  = CAT + (size_t)NTOK * DCAT;        // NTOK*256
  float* Rb  = Zb  + (size_t)NTOK * HDIM;
  float* Qb  = Rb  + (size_t)NTOK * HDIM;
  float* Kb  = Qb  + (size_t)NTOK * HDIM;
  float* Vb  = Kb  + (size_t)NTOK * HDIM;
  float* Hh  = Vb  + (size_t)NTOK * HDIM;        // attn out / att-block out
  float* Xb  = Hh  + (size_t)NTOK * HDIM;        // ln1 out
  float* FF  = Xb  + (size_t)NTOK * HDIM;        // NTOK*1024
  float* Ob  = FF  + (size_t)NTOK * FFDIM;       // ff2 out
  float* EH  = Ob  + (size_t)NTOK * HDIM;        // proj out

  auto gemm = [&](const float* A, const float* W, const float* bias, float* C,
                  int N, int K, int lda, int segStride, int rowOff, int act) {
    dim3 grid(N / 128, NTOK / 128);
    wmma_gemm_kernel<<<grid, 256, 0, stream>>>(A, W, bias, C, NTOK, N, K,
                                               lda, segStride, rowOff, act);
  };
  const int CAT_BLKS = (NTOK * DCAT) / 256;      // 49152
  const int ROW_BLKS = NTOK;                     // 32768

  for (int phase = 0; phase < 2; ++phase) {
    const int srow = phase ? 512 : 0;
    const int gz_w = phase ? GUL_W : GUE_W, gz_b = phase ? GUL_B : GUE_B;
    const int gr_w = phase ? GRL_W : GRE_W, gr_b = phase ? GRL_B : GRE_B;
    const int base = phase ? AL_QW : AE_QW;      // att param block base
    const int pj_w = phase ? PHL_W : PHE_W, pj_b = phase ? PHL_B : PHE_B;

    // cat = [ix | hx] ; z = sigmoid(cat@Wz+bz) ; r = sigmoid(cat@Wr+br)
    cat_kernel<<<CAT_BLKS, 256, 0, stream>>>(input_, hx, nullptr, CAT, srow);
    gemm(CAT, P[gz_w], P[gz_b], Zb, HDIM, DCAT, DCAT, 512, 0, 1);
    gemm(CAT, P[gr_w], P[gr_b], Rb, HDIM, DCAT, DCAT, 512, 0, 1);
    // x2 = [ix | r*hx]  (overwrites CAT)
    cat_kernel<<<CAT_BLKS, 256, 0, stream>>>(input_, hx, Rb, CAT, srow);

    // QKV projections
    gemm(CAT, P[base + 0], P[base + 1], Qb, HDIM, DCAT, DCAT, 512, 0, 0);
    if (phase == 0) {   // self-attention on x2
      gemm(CAT, P[AE_KW], P[AE_KB], Kb, HDIM, DCAT, DCAT, 512, 0, 0);
      gemm(CAT, P[AE_VW], P[AE_VB], Vb, HDIM, DCAT, DCAT, 512, 0, 0);
    } else {            // cross-attention: K/V from eh_next (entity half of d_out)
      gemm(out, P[AL_KW], P[AL_KB], Kb, HDIM, HDIM, HDIM, 1024, 0, 0);
      gemm(out, P[AL_VW], P[AL_VB], Vb, HDIM, HDIM, HDIM, 1024, 0, 0);
    }

    // fused multihead attention
    attn_kernel<<<dim3(SEQ / 128, NHEADS, BATCH), 256, 0, stream>>>(Qb, Kb, Vb, Hh);

    // x = LN(h + pq) ; ff = tanh(x@W1+b1)@W2+b2 ; out_blk = LN(ff + x)
    add_ln_kernel<<<ROW_BLKS, 256, 0, stream>>>(Hh, Qb, P[base + 10], P[base + 11], Xb);
    gemm(Xb, P[base + 6], P[base + 7], FF, FFDIM, HDIM, HDIM, 512, 0, 2);
    gemm(FF, P[base + 8], P[base + 9], Ob, HDIM, FFDIM, FFDIM, 512, 0, 0);
    add_ln_kernel<<<ROW_BLKS, 256, 0, stream>>>(Ob, Xb, P[base + 12], P[base + 13], Hh);

    // h = tanh(blk @ Wp + bp) ; next = (1-z)*hx + z*h  -> d_out slice
    gemm(Hh, P[pj_w], P[pj_b], EH, HDIM, HDIM, HDIM, 512, 0, 2);
    gru_kernel<<<ROW_BLKS, 256, 0, stream>>>(Zb, EH, hx, out, srow);
  }
}
